// Attention_65747359367242
// MI455X (gfx1250) — matrix operand order repaired and checked by reading, not compiled, not run on hardware
//
#include <hip/hip_runtime.h>
#include <hip/hip_bf16.h>

// ---------------------------------------------------------------------------
// Attention (tanh, head-mean) for MI455X / gfx1250, wave32 + WMMA bf16 + TDM.
//   B=8, N=2048, D=128, H=4, DH=32
// Phase 1: QKV projection (bf16 WMMA GEMM) -> Qb, Kb row-major bf16; Vt d-major.
// Phase 2: per-wave 16-row slab; St = K*Q^T per head via WMMA, tanh+mean,
//          score registers feed directly (no cross-lane moves) into O = A*V.
//          K tile staged via TDM tensor_load_to_lds (TENSORcnt), V tile via
//          global_load_async_to_lds_b128 (ASYNCcnt), double-buffered in LDS.
// ---------------------------------------------------------------------------

#define B_  8
#define N_  2048
#define D_  128
#define H_  4
#define DH_ 32
#define SCALE_ 0.08838834764831845f   // 1/sqrt(128)

typedef __attribute__((ext_vector_type(16))) __bf16 v16bf;
typedef __attribute__((ext_vector_type(8)))  __bf16 v8bf;
typedef __attribute__((ext_vector_type(8)))  float  v8f;
typedef __attribute__((ext_vector_type(4)))  float  v4f;
typedef __attribute__((ext_vector_type(4)))  unsigned int u32x4;
typedef __attribute__((ext_vector_type(8)))  unsigned int u32x8;

#if __has_builtin(__builtin_amdgcn_tanhf)
#define TANHF(x) __builtin_amdgcn_tanhf(x)
#else
#define TANHF(x) tanhf(x)
#endif

#if __has_builtin(__builtin_amdgcn_global_load_async_to_lds_b128)
#define HAVE_ASYNC 1
#else
#define HAVE_ASYNC 0
#endif

// Builtin takes pointer-to-v4i in global (AS1) / LDS (AS3) address spaces.
typedef int v4i_ __attribute__((vector_size(16)));
typedef __attribute__((address_space(1))) v4i_* gv4i_ptr;
typedef __attribute__((address_space(3))) v4i_* lv4i_ptr;

__device__ __forceinline__ lv4i_ptr to_lds(void* p) {
  // flat -> LDS address space is a low-32-bit truncation on amdgcn
  return (lv4i_ptr)(unsigned int)(unsigned long long)p;
}
__device__ __forceinline__ gv4i_ptr to_glob(const void* p) {
  return (gv4i_ptr)(unsigned long long)p;
}

__device__ __forceinline__ void wait_async0() {
#if __has_builtin(__builtin_amdgcn_s_wait_asynccnt)
  __builtin_amdgcn_s_wait_asynccnt(0);
#else
  asm volatile("s_wait_asynccnt 0x0" ::: "memory");
#endif
}

// ---------------------------------------------------------------------------
// Phase 1: Q = x@Wq+bq, K = cond@Wk+bk (row-major bf16), V = cond@Wv+bv
// stored transposed: Vt[b][d][n] bf16.  One wave per 16x16 output tile.
// ---------------------------------------------------------------------------
__global__ __launch_bounds__(32) void qkv_proj_kernel(
    const float* __restrict__ x, const float* __restrict__ cond,
    const float* __restrict__ Wq, const float* __restrict__ bq,
    const float* __restrict__ Wk, const float* __restrict__ bk,
    const float* __restrict__ Wv, const float* __restrict__ bv,
    __bf16* __restrict__ Qb, __bf16* __restrict__ Kb, __bf16* __restrict__ Vt)
{
  const int lane = threadIdx.x & 31;
  const int half = lane >> 4;
  const int l15  = lane & 15;
  const int rt = blockIdx.x;          // 16-row tile over B*N (0..1023)
  const int ct = blockIdx.y;          // 16-col tile over 128 (0..7)
  const int pj = blockIdx.z;          // 0:Q 1:K 2:V
  const int r0 = rt * 16, c0 = ct * 16;

  const float* src  = (pj == 0) ? x  : cond;
  const float* W    = (pj == 0) ? Wq : (pj == 1) ? Wk : Wv;
  const float* bias = (pj == 0) ? bq : (pj == 1) ? bk : bv;

  v8f acc = {0.f, 0.f, 0.f, 0.f, 0.f, 0.f, 0.f, 0.f};
  const float* arow = src + (size_t)(r0 + l15) * D_;

#pragma unroll
  for (int kc = 0; kc < 4; ++kc) {
    // A fragment (16x32 bf16): row M = lane&15; elem e<8 -> k=e+8*half,
    // e>=8 -> k=16+(e-8)+8*half  (within this 32-wide k chunk).
    const float* ap = arow + kc * 32 + 8 * half;
    v4f a0 = *(const v4f*)(ap);
    v4f a1 = *(const v4f*)(ap + 4);
    v4f a2 = *(const v4f*)(ap + 16);
    v4f a3 = *(const v4f*)(ap + 20);
    v16bf a;
#pragma unroll
    for (int e = 0; e < 4; ++e) {
      a[e]      = (__bf16)a0[e];
      a[4 + e]  = (__bf16)a1[e];
      a[8 + e]  = (__bf16)a2[e];
      a[12 + e] = (__bf16)a3[e];
    }
    // B fragment (32x16 bf16): col = lane&15; elem e<8 -> k = 8*half + e,
    // e>=8 -> k = 16 + 8*half + (e-8).
    v16bf bm;
    const float* wp = W + (size_t)(kc * 32 + 8 * half) * D_ + c0 + l15;
#pragma unroll
    for (int e = 0; e < 16; ++e) bm[e] = (__bf16)wp[(size_t)(e + (e >= 8 ? 8 : 0)) * D_];

    acc = __builtin_amdgcn_wmma_f32_16x16x32_bf16(false, a, false, bm,
                                                  (short)0, acc, false, false);
  }

  const float badd = bias[c0 + l15];
#pragma unroll
  for (int i = 0; i < 8; ++i) acc[i] += badd;

  // C/D layout: element i -> row = i + 8*half, col = lane&15.
#pragma unroll
  for (int i = 0; i < 8; ++i) {
    const int row = r0 + i + 8 * half;
    const int col = c0 + l15;
    const __bf16 v = (__bf16)acc[i];
    if (pj == 0)      Qb[(size_t)row * D_ + col] = v;
    else if (pj == 1) Kb[(size_t)row * D_ + col] = v;
    else {
      const int bb = row >> 11, nn = row & (N_ - 1);
      Vt[((size_t)bb * D_ + col) * N_ + nn] = v;
    }
  }
}

// ---------------------------------------------------------------------------
// Phase 2 attention kernel.
// ---------------------------------------------------------------------------
__global__ __launch_bounds__(256) void attn_kernel(
    const __bf16* __restrict__ Qb, const __bf16* __restrict__ Kb,
    const __bf16* __restrict__ Vt, const float* __restrict__ mask,
    float* __restrict__ out)
{
  __shared__ __align__(128) __bf16 skK[2][32 * D_];   // [m 0..31][d 0..127]
  __shared__ __align__(128) __bf16 skV[2][D_ * 32];   // [d 0..127][m 0..31]

  const int tid  = threadIdx.x;
  const int lane = tid & 31;
  const int wv   = tid >> 5;            // wave in block (0..7)
  const int half = lane >> 4;
  const int l15  = lane & 15;

  const int b  = blockIdx.x >> 4;                       // 16 blocks / batch
  const int n0 = ((blockIdx.x & 15) * 8 + wv) * 16;

  const int vd = tid >> 1, vh = tid & 1;                // V-tile copy mapping

  // --- K tile: one contiguous 8 KiB block -> TDM 1-D descriptor (8B units).
  auto issue_K = [&](int mc, int bu) {
    const unsigned long long ga =
        (unsigned long long)(Kb + ((size_t)b * N_ + (size_t)mc * 32) * D_);
    const unsigned int la = (unsigned int)(unsigned long long)(void*)&skK[bu][0];
    u32x4 g0;
    g0[0] = 1u;                                        // count=1, user mode
    g0[1] = la;                                        // lds_addr (bytes)
    g0[2] = (unsigned int)(ga & 0xFFFFFFFFull);        // global_addr[31:0]
    g0[3] = ((unsigned int)(ga >> 32) & 0x01FFFFFFu)   // global_addr[56:32]
            | (2u << 30);                              // type = 2 (image)
    const unsigned int KE = 1024u;                     // 8 KiB / 8B
    u32x8 g1;
    g1[0] = 3u << 16;                                  // data_size = 8 bytes
    g1[1] = (KE & 0xFFFFu) << 16;                      // tensor_dim0[15:0]
    g1[2] = (KE >> 16) | (1u << 16);                   // dim0[31:16] | tensor_dim1=1
    g1[3] = (KE & 0xFFFFu) << 16;                      // tile_dim0 = 1024
    g1[4] = 0u;                                        // tile_dim1/2 unused
    g1[5] = KE;                                        // tensor_dim0_stride lo32
    g1[6] = 0u;
    g1[7] = 0u;
    asm volatile("tensor_load_to_lds %0, %1" :: "s"(g0), "s"(g1) : "memory");
  };

  // --- V tile: 128 rows x 64B from Vt[b][d][m0..m0+31] -> async to LDS.
  auto issue_V = [&](int mc, int bu) {
    const __bf16* gv = Vt + ((size_t)b * D_ + vd) * N_ + (size_t)mc * 32 + vh * 16;
    __bf16* lv = &skV[bu][vd * 32 + vh * 16];
#if HAVE_ASYNC
    __builtin_amdgcn_global_load_async_to_lds_b128(to_glob(gv),     to_lds(lv),     0, 0);
    __builtin_amdgcn_global_load_async_to_lds_b128(to_glob(gv + 8), to_lds(lv + 8), 0, 0);
#else
    uint4 t0 = ((const uint4*)gv)[0];
    uint4 t1 = ((const uint4*)gv)[1];
    ((uint4*)lv)[0] = t0;
    ((uint4*)lv)[1] = t1;
#endif
  };

  auto wait_tiles = [&]() {
    if (wv == 0) __builtin_amdgcn_s_wait_tensorcnt(0);   // K tile (TDM)
    wait_async0();                                       // V tile (async)
  };

  // Q fragments: B-operand of S-GEMM (col n = lane&15, k=d contiguous) — fixed.
  v16bf qf[H_];
#pragma unroll
  for (int h = 0; h < H_; ++h)
    qf[h] = *(const v16bf*)(Qb + ((size_t)b * N_ + n0 + l15) * D_ + h * DH_ + 16 * half);

  const v8f z8 = {0.f, 0.f, 0.f, 0.f, 0.f, 0.f, 0.f, 0.f};
  v8f o[8];
#pragma unroll
  for (int dt = 0; dt < 8; ++dt) o[dt] = z8;

  if (wv == 0) issue_K(0, 0);
  issue_V(0, 0);
  wait_tiles();
  __syncthreads();

  const float* mrow = mask + ((size_t)b * N_ + n0 + l15) * (size_t)N_;

  for (int mc = 0; mc < N_ / 32; ++mc) {
    const int cur = mc & 1;
    const int m0  = mc * 32;
    const bool more = (mc + 1) < (N_ / 32);
    if (more) {                       // transfers into other buffer overlap compute
      if (wv == 0) issue_K(mc + 1, cur ^ 1);
      issue_V(mc + 1, cur ^ 1);
    }

    // mask is streamed exactly once -> non-temporal (preserve L2 for Q/K/V)
    const float* mp = mrow + m0 + 8 * half;
    v4f mk00 = __builtin_nontemporal_load((const v4f*)(mp));
    v4f mk01 = __builtin_nontemporal_load((const v4f*)(mp + 4));
    v4f mk10 = __builtin_nontemporal_load((const v4f*)(mp + 16));
    v4f mk11 = __builtin_nontemporal_load((const v4f*)(mp + 20));
    __builtin_prefetch(mp + 32, 0, 0);  // next mask chunk -> global_prefetch

    float as0[8], as1[8];
#pragma unroll
    for (int i = 0; i < 8; ++i) { as0[i] = 0.f; as1[i] = 0.f; }

#pragma unroll
    for (int h = 0; h < H_; ++h) {
      // A-operand = K rows (m in lane&15), k = d within head (2x 16B chunks)
      const __bf16* kp0 = &skK[cur][(0  + l15) * D_ + h * DH_ + 16 * half];
      const __bf16* kp1 = &skK[cur][(16 + l15) * D_ + h * DH_ + 16 * half];
      v8bf k0l = *(const v8bf*)kp0;
      v8bf k0h = *(const v8bf*)(kp0 + 8);
      v8bf k1l = *(const v8bf*)kp1;
      v8bf k1h = *(const v8bf*)(kp1 + 8);
      v16bf k0, k1;
#pragma unroll
      for (int e = 0; e < 8; ++e) {
        k0[e] = k0l[e]; k0[8 + e] = k0h[e];
        k1[e] = k1l[e]; k1[8 + e] = k1h[e];
      }
      // St[m][n] = sum_d K[m,d]*Q[n,d]  (n in lanes, m in acc registers)
      v8f s0 = __builtin_amdgcn_wmma_f32_16x16x32_bf16(false, k0, false, qf[h],
                                                       (short)0, z8, false, false);
      v8f s1 = __builtin_amdgcn_wmma_f32_16x16x32_bf16(false, k1, false, qf[h],
                                                       (short)0, z8, false, false);
#pragma unroll
      for (int i = 0; i < 4; ++i) {
        as0[i]     += TANHF(fmaf(s0[i],     SCALE_, mk00[i]));
        as0[4 + i] += TANHF(fmaf(s0[4 + i], SCALE_, mk01[i]));
        as1[i]     += TANHF(fmaf(s1[i],     SCALE_, mk10[i]));
        as1[4 + i] += TANHF(fmaf(s1[4 + i], SCALE_, mk11[i]));
      }
    }

    // Score tile registers map 1:1 onto the next A-fragment (no shuffles):
    // elem e<8 -> m = m0 + e + 8*half (as0[e]); e>=8 -> m0+16+(e-8)+8*half.
    v16bf af;
#pragma unroll
    for (int e = 0; e < 8; ++e) {
      af[e]     = (__bf16)(as0[e] * 0.25f);
      af[8 + e] = (__bf16)(as1[e] * 0.25f);
    }

#pragma unroll
    for (int dt = 0; dt < 8; ++dt) {
      // B-operand = V (e<8 -> k = m 8*half+e, e>=8 -> 16+8*half+(e-8); col d = lane&15)
      const __bf16* vp = &skV[cur][(dt * 16 + l15) * 32 + 8 * half];
      v8bf vl = *(const v8bf*)vp;
      v8bf vh8 = *(const v8bf*)(vp + 16);
      v16bf vf;
#pragma unroll
      for (int e = 0; e < 8; ++e) { vf[e] = vl[e]; vf[8 + e] = vh8[e]; }
      o[dt] = __builtin_amdgcn_wmma_f32_16x16x32_bf16(false, af, false, vf,
                                                      (short)0, o[dt], false, false);
    }

    if (more) wait_tiles();           // next buffer ready before the barrier
    __syncthreads();
  }

  // O tile: element i -> row n = n0 + i + 8*half, col d = dt*16 + lane&15
#pragma unroll
  for (int dt = 0; dt < 8; ++dt) {
#pragma unroll
    for (int i = 0; i < 8; ++i) {
      float* op = out + ((size_t)b * N_ + n0 + i + 8 * half) * D_ + dt * 16 + l15;
      __builtin_nontemporal_store(o[dt][i], op);
    }
  }
}

// ---------------------------------------------------------------------------
extern "C" void kernel_launch(void* const* d_in, const int* in_sizes, int n_in,
                              void* d_out, int out_size, void* d_ws, size_t ws_size,
                              hipStream_t stream) {
  (void)in_sizes; (void)n_in; (void)out_size; (void)ws_size;
  const float* x    = (const float*)d_in[0];
  const float* cond = (const float*)d_in[1];
  /* d_in[2] = flags, unused by the reference forward */
  const float* mask = (const float*)d_in[3];
  const float* Wq   = (const float*)d_in[4];
  const float* bq   = (const float*)d_in[5];
  const float* Wk   = (const float*)d_in[6];
  const float* bk   = (const float*)d_in[7];
  const float* Wv   = (const float*)d_in[8];
  const float* bv   = (const float*)d_in[9];

  char* ws = (char*)d_ws;
  const size_t SZ = (size_t)B_ * N_ * D_ * sizeof(__bf16);  // 4 MiB each
  __bf16* Qb = (__bf16*)(ws);
  __bf16* Kb = (__bf16*)(ws + SZ);
  __bf16* Vt = (__bf16*)(ws + 2 * SZ);

  qkv_proj_kernel<<<dim3((B_ * N_) / 16, D_ / 16, 3), 32, 0, stream>>>(
      x, cond, Wq, bq, Wk, bk, Wv, bv, Qb, Kb, Vt);
  attn_kernel<<<dim3(B_ * (N_ / 128)), 256, 0, stream>>>(
      Qb, Kb, Vt, mask, (float*)d_out);
}
